// GlobalPointer_48077863911943
// MI455X (gfx1250) — matrix operand order repaired and checked by reading, not compiled, not run on hardware
//
#include <hip/hip_runtime.h>
#include <hip/hip_bf16.h>
#include <math.h>

typedef _Float16 f16;
typedef __attribute__((ext_vector_type(16))) _Float16 v16h;
typedef __attribute__((ext_vector_type(8)))  _Float16 v8h;
typedef __attribute__((ext_vector_type(8)))  float    v8f;

#define HEADS   12
#define HEAD    64
#define BATCH   8
#define SEQ     1024
#define HID     768
#define NOUT    1536            // HEADS*HEAD*2
#define MROWS   (BATCH*SEQ)     // 8192
#define NEG_INF (-1000000000000.0f)

// ---- workspace layout (bytes) ----
#define XH_OFF   ((size_t)0)                         // 8192*768  f16 = 12,582,912
#define WT_OFF   ((size_t)12582912)                  // 1536*768  f16 =  2,359,296
#define QH_OFF   ((size_t)(12582912 + 2359296))      // 96*1024*64 f16 = 12,582,912
#define KH_OFF   ((size_t)(12582912 + 2359296 + 12582912))

__device__ __forceinline__ v16h load_frag(const f16* lo, const f16* hi) {
  v8h a = *(const v8h*)lo;
  v8h b = *(const v8h*)hi;
  return __builtin_shufflevector(a, b, 0,1,2,3,4,5,6,7,8,9,10,11,12,13,14,15);
}

// ---------------- converts (single-use reads -> NT loads) ----------------
__global__ void cvt_x_kernel(const float* __restrict__ X, f16* __restrict__ Xh, int n) {
  int i = blockIdx.x * blockDim.x + threadIdx.x;
  if (i < n) Xh[i] = (f16)__builtin_nontemporal_load(X + i);
}

__global__ void cvt_wT_kernel(const float* __restrict__ W, f16* __restrict__ WT) {
  int i = blockIdx.x * blockDim.x + threadIdx.x;   // over 1536*768
  if (i < NOUT * HID) {
    int n = i / HID, k = i - n * HID;
    WT[i] = (f16)__builtin_nontemporal_load(W + (size_t)k * NOUT + n);  // WT[n][k] = W[k][n]
  }
}

// ---------------- GEMM1: x = Xh @ W + b, fused RoPE, split Q/K ----------------
// block: 256 threads = 8 waves, block tile 64(M) x 128(N), wave tile 32x32
__global__ __launch_bounds__(256) void gemm_qkv_rope_kernel(
    const f16* __restrict__ Xh, const f16* __restrict__ WT,
    const float* __restrict__ bias,
    f16* __restrict__ Qh, f16* __restrict__ Kh)
{
  const int lane  = threadIdx.x & 31;
  const int wave  = threadIdx.x >> 5;
  const int lg    = lane >> 4;           // lane group 0/1
  const int lc    = lane & 15;
  const int waveM = wave >> 2;           // 0..1
  const int waveN = wave & 3;            // 0..3
  const int rowBase = blockIdx.y * 64 + waveM * 32;
  const int colBase = blockIdx.x * 128 + waveN * 32;

  v8f acc[2][2] = {};

  for (int k0 = 0; k0 < HID; k0 += 32) {
    v16h aF[2], bF[2];
#pragma unroll
    for (int mt = 0; mt < 2; ++mt) {
      const f16* base = Xh + (size_t)(rowBase + mt * 16 + lc) * HID + k0 + lg * 8;
      aF[mt] = load_frag(base, base + 16);     // K = lg*8+{0..7}, 16+lg*8+{0..7}
    }
#pragma unroll
    for (int nt = 0; nt < 2; ++nt) {
      const f16* base = WT + (size_t)(colBase + nt * 16 + lc) * HID + k0 + lg * 8;
      bF[nt] = load_frag(base, base + 16);     // K = lg*8+{0..7}, 16+lg*8+{0..7}
    }
#pragma unroll
    for (int mt = 0; mt < 2; ++mt)
#pragma unroll
      for (int nt = 0; nt < 2; ++nt)
        acc[mt][nt] = __builtin_amdgcn_wmma_f32_16x16x32_f16(
            false, aF[mt], false, bF[nt], (short)0, acc[mt][nt], false, false);
  }

  // epilogue: + bias, RoPE rotate, write f16 Q / K as [b][h][s][d]
#pragma unroll
  for (int mt = 0; mt < 2; ++mt) {
#pragma unroll
    for (int nt = 0; nt < 2; ++nt) {
      const int j = colBase + nt * 16 + lc;     // output column 0..1535
      const int h = j >> 7;                     // head
      const int c = j & 127;                    // channel within head pair (q|k)
      const int d = c & 63;                     // dim within q or k
      const int freq = d & 31;
      // 10000^{-2*freq/64} = exp2(-freq * (2/64)*log2(10000))
      const float inv = exp2f((float)freq * -0.41524101187f);
      const float bj  = bias[j];
      f16* dst = (c < HEAD) ? Qh : Kh;
#pragma unroll
      for (int r = 0; r < 8; ++r) {
        const int m = rowBase + mt * 16 + lg * 8 + r;   // global row
        const int s = m & (SEQ - 1);                    // position in sequence
        const int b = m >> 10;                          // batch
        float val  = acc[mt][nt][r] + bj;
        float part = __shfl_xor(val, 1, 32);            // partner channel d^1 (has its own bias)
        float ang  = (float)s * inv;
        float sn, cs;
        __sincosf(ang, &sn, &cs);
        float t2   = (d & 1) ? part : -part;            // even d: -t[d+1]; odd d: +t[d-1]
        float rot  = val * cs + t2 * sn;
        dst[((size_t)((b * HEADS + h) * SEQ + s)) * HEAD + d] = (f16)rot;
      }
    }
  }
}

// ---------------- GEMM2: logits[b,h] = Q @ K^T, fused mask/causal/scale ----------------
// block: 256 threads = 8 waves, block tile 64(M) x 128(N) per (b,h) plane.
// Output stream (384 MB) is 2x the 192 MB L2 and never re-read: store TH_NT
// so it does not evict the L2-resident Q/K working set.
__global__ __launch_bounds__(256) void attn_logits_kernel(
    const f16* __restrict__ Qh, const f16* __restrict__ Kh,
    const float* __restrict__ mask, float* __restrict__ out)
{
  const int lane  = threadIdx.x & 31;
  const int wave  = threadIdx.x >> 5;
  const int lg    = lane >> 4;
  const int lc    = lane & 15;
  const int waveM = wave >> 2;
  const int waveN = wave & 3;
  const int p = blockIdx.z;              // plane = b*HEADS + h
  const int b = p / HEADS;
  const int mBase = blockIdx.y * 64 + waveM * 32;
  const int nBase = blockIdx.x * 128 + waveN * 32;
  const size_t plane = (size_t)p * SEQ * HEAD;

  v8f acc[2][2] = {};
#pragma unroll
  for (int k0 = 0; k0 < HEAD; k0 += 32) {
    v16h aF[2], bF[2];
#pragma unroll
    for (int mt = 0; mt < 2; ++mt) {
      const f16* base = Qh + plane + (size_t)(mBase + mt * 16 + lc) * HEAD + k0 + lg * 8;
      aF[mt] = load_frag(base, base + 16);
    }
#pragma unroll
    for (int nt = 0; nt < 2; ++nt) {
      const f16* base = Kh + plane + (size_t)(nBase + nt * 16 + lc) * HEAD + k0 + lg * 8;
      bF[nt] = load_frag(base, base + 16);
    }
#pragma unroll
    for (int mt = 0; mt < 2; ++mt)
#pragma unroll
      for (int nt = 0; nt < 2; ++nt)
        acc[mt][nt] = __builtin_amdgcn_wmma_f32_16x16x32_f16(
            false, aF[mt], false, bF[nt], (short)0, acc[mt][nt], false, false);
  }

  // epilogue: mask rows/cols, strict-lower-tri -1e12, scale 1/sqrt(64)
#pragma unroll
  for (int mt = 0; mt < 2; ++mt) {
#pragma unroll
    for (int nt = 0; nt < 2; ++nt) {
      const int n  = nBase + nt * 16 + lc;
      const float mc = mask[b * SEQ + n];
#pragma unroll
      for (int r = 0; r < 8; ++r) {
        const int m  = mBase + mt * 16 + lg * 8 + r;
        const float mr = mask[b * SEQ + m];
        float L = acc[mt][nt][r];
        L = L * mr + NEG_INF * (1.0f - mr);
        L = L * mc + NEG_INF * (1.0f - mc);
        if (m > n) L -= 1.0e12f;
        __builtin_nontemporal_store(L * 0.125f,
            out + ((size_t)p * SEQ + m) * SEQ + n);
      }
    }
  }
}

extern "C" void kernel_launch(void* const* d_in, const int* in_sizes, int n_in,
                              void* d_out, int out_size, void* d_ws, size_t ws_size,
                              hipStream_t stream) {
  const float* X    = (const float*)d_in[0];   // (8,1024,768)
  const float* mask = (const float*)d_in[1];   // (8,1024)
  const float* W    = (const float*)d_in[2];   // (768,1536)
  const float* bias = (const float*)d_in[3];   // (1536,)
  float* out = (float*)d_out;                  // (8,12,1024,1024)

  char* ws = (char*)d_ws;
  f16* Xh = (f16*)(ws + XH_OFF);
  f16* WT = (f16*)(ws + WT_OFF);
  f16* Qh = (f16*)(ws + QH_OFF);
  f16* Kh = (f16*)(ws + KH_OFF);

  {
    int n = MROWS * HID;
    cvt_x_kernel<<<(n + 255) / 256, 256, 0, stream>>>(X, Xh, n);
  }
  {
    int n = NOUT * HID;
    cvt_wT_kernel<<<(n + 255) / 256, 256, 0, stream>>>(W, WT);
  }
  {
    dim3 grid(NOUT / 128, MROWS / 64, 1);      // (12, 128)
    gemm_qkv_rope_kernel<<<grid, 256, 0, stream>>>(Xh, WT, bias, Qh, Kh);
  }
  {
    dim3 grid(SEQ / 128, SEQ / 64, BATCH * HEADS);  // (8, 16, 96)
    attn_logits_kernel<<<grid, 256, 0, stream>>>(Qh, Kh, mask, out);
  }
}
